// GCNNet2_5781025980783
// MI455X (gfx1250) — matrix operand order repaired and checked by reading, not compiled, not run on hardware
//
#include <hip/hip_runtime.h>
#include <hip/hip_bf16.h>
#include <math.h>

typedef __attribute__((ext_vector_type(16))) __bf16 v16bf;
typedef __attribute__((ext_vector_type(8)))  float  v8f;

#define NNODES 100000
#define FIN    128
#define HID    256
#define FOUT   40
#define FOUTP  48   // padded to 3 x 16 WMMA column tiles

__device__ __forceinline__ __bf16 f2bf(float f) { return (__bf16)f; }

// ---------------- utility ----------------
__global__ void zero_f32_kernel(float* p, long long n) {
    long long i = (long long)blockIdx.x * blockDim.x + threadIdx.x;
    if (i < n) p[i] = 0.0f;
}

__global__ void degree_kernel(const long long* __restrict__ dst, float* deg, int E) {
    int i = blockIdx.x * blockDim.x + threadIdx.x;
    if (i < E) atomicAdd(&deg[(int)dst[i]], 1.0f);
}

__global__ void rsqrt_kernel(float* deg, int n) {
    int i = blockIdx.x * blockDim.x + threadIdx.x;
    if (i < n) deg[i] = rsqrtf(deg[i] + 1.0f);
}

// ---------------- pack W[K, ncolReal] (f32 row-major) -> Wt[ncolPad, K] (bf16, zero pad) ----------------
__global__ void pack_wt_kernel(const float* __restrict__ W, __bf16* __restrict__ Wt,
                               int K, int ncolReal, int ncolPad) {
    int idx = blockIdx.x * blockDim.x + threadIdx.x;
    if (idx >= ncolPad * K) return;
    int c = idx / K;
    int k = idx % K;
    float v = (c < ncolReal) ? W[(size_t)k * ncolReal + c] : 0.0f;
    Wt[idx] = f2bf(v);
}

// ---------------- WMMA GEMM:  Y[nrows, ncolPad] = X[nrows, K] @ Wt^T ----------------
// One wave computes one 16x16 output tile; fully unrolled K loop of
// v_wmma_f32_16x16x32_bf16. B fragments are contiguous 32B loads from packed Wt.
template <int K>
__global__ void __launch_bounds__(256)
gemm_bf16_wmma_kernel(const float* __restrict__ X,
                      const __bf16* __restrict__ Wt,   // [ncolPad][K] bf16, pad cols zeroed
                      float* __restrict__ Y,
                      int nrows, int ncolPad) {
    const int wavesPerBlock = blockDim.x >> 5;
    const int waveId  = blockIdx.x * wavesPerBlock + (threadIdx.x >> 5);
    const int nColT   = ncolPad >> 4;
    const int nRowT   = nrows >> 4;
    if (waveId >= nRowT * nColT) return;              // wave-uniform: EXEC all-ones inside

    const int rowBase = (waveId / nColT) << 4;
    const int colBase = (waveId % nColT) << 4;

    const int lane = threadIdx.x & 31;
    const int hl   = lane >> 4;       // half-wave select
    const int l16  = lane & 15;

    const float*  xrow = X  + (size_t)(rowBase + l16) * K;        // A: my row
    const __bf16* wcol = Wt + (size_t)(colBase + l16) * K + hl * 16;  // B: my column strip

    v8f acc = {};

#pragma unroll
    for (int k0 = 0; k0 < K; k0 += 32) {
        // ---- A tile 16x32 bf16 (row-major f32 X) contiguous: element s holds k = 16*hl + s ----
        v16bf a;
#pragma unroll
        for (int i = 0; i < 4; ++i) {
            int kk = k0 + hl * 16 + 2 * i;            // lanes 0-15: K 0..15 ; lanes 16-31: K 16..31
            a[2 * i]     = f2bf(xrow[kk]);
            a[2 * i + 1] = f2bf(xrow[kk + 1]);
            a[2 * i + 8] = f2bf(xrow[kk + 8]);        // VGPR4..7: +8 in K
            a[2 * i + 9] = f2bf(xrow[kk + 9]);
        }
        // ---- B tile 32x16 bf16: 16 consecutive K values, contiguous in packed Wt ----
        v16bf b = *(const v16bf*)(wcol + k0);         // 32B -> 2x global_load_b128

        acc = __builtin_amdgcn_wmma_f32_16x16x32_bf16(
            /*neg_a=*/false, a, /*neg_b=*/false, b,
            /*c_mod=*/(short)0, acc, /*reuse_a=*/false, /*reuse_b=*/false);
    }

    // ---- store C: VGPR i -> M = i + (hl?8:0), N = l16 ----
#pragma unroll
    for (int i = 0; i < 8; ++i) {
        int row = rowBase + (hl ? 8 : 0) + i;
        Y[(size_t)row * ncolPad + colBase + l16] = acc[i];
    }
}

// ---------------- edge gather * norm -> scatter-add (float4 per thread) ----------------
__global__ void scatter4_kernel(const long long* __restrict__ src,
                                const long long* __restrict__ dst,
                                const float* __restrict__ dis,
                                const float* __restrict__ xw, int ldxw,
                                float* __restrict__ agg, int ldagg,
                                int ncol4, int E) {
    long long gid = (long long)blockIdx.x * blockDim.x + threadIdx.x;
    long long total = (long long)E * ncol4;
    if (gid >= total) return;
    int e  = (int)(gid / ncol4);
    int f4 = (int)(gid % ncol4) << 2;
    int s = (int)src[e];
    int d = (int)dst[e];
    float norm = dis[s] * dis[d];
    const float4 v = *(const float4*)(xw + (size_t)s * ldxw + f4);
    float* ap = agg + (size_t)d * ldagg + f4;
    atomicAdd(ap + 0, v.x * norm);
    atomicAdd(ap + 1, v.y * norm);
    atomicAdd(ap + 2, v.z * norm);
    atomicAdd(ap + 3, v.w * norm);
}

// ---------------- fused: h = relu(agg + dis^2 * xw + b)  (in place on agg) ----------------
__global__ void fuse_relu_kernel(float* __restrict__ agg,
                                 const float* __restrict__ xw,
                                 const float* __restrict__ dis,
                                 const float* __restrict__ bias,
                                 int nrows, int ncol) {
    long long i = (long long)blockIdx.x * blockDim.x + threadIdx.x;
    long long total = (long long)nrows * ncol;
    if (i >= total) return;
    int n = (int)(i / ncol);
    int f = (int)(i % ncol);
    float dd = dis[n] * dis[n];
    float v = agg[i] + dd * xw[i] + bias[f];
    agg[i] = fmaxf(v, 0.0f);
}

// ---------------- fused: out = log_softmax(agg2 + dis^2 * xw2 + b2) ----------------
__global__ void final_kernel(const float* __restrict__ agg2,       // [N, 40]
                             const float* __restrict__ xw2,        // [N, 48] padded
                             const float* __restrict__ dis,
                             const float* __restrict__ b2,
                             float* __restrict__ out, int nrows) {
    int n = blockIdx.x * blockDim.x + threadIdx.x;
    if (n >= nrows) return;
    float dd = dis[n] * dis[n];
    float v[FOUT];
    float mx = -INFINITY;
#pragma unroll
    for (int f = 0; f < FOUT; ++f) {
        v[f] = agg2[(size_t)n * FOUT + f] + dd * xw2[(size_t)n * FOUTP + f] + b2[f];
        mx = fmaxf(mx, v[f]);
    }
    float s = 0.0f;
#pragma unroll
    for (int f = 0; f < FOUT; ++f) s += __expf(v[f] - mx);
    float lse = mx + __logf(s);
#pragma unroll
    for (int f = 0; f < FOUT; ++f) out[(size_t)n * FOUT + f] = v[f] - lse;
}

static inline size_t align_up(size_t x, size_t a) { return (x + a - 1) & ~(a - 1); }

extern "C" void kernel_launch(void* const* d_in, const int* in_sizes, int n_in,
                              void* d_out, int out_size, void* d_ws, size_t ws_size,
                              hipStream_t stream) {
    const float*     x   = (const float*)d_in[0];
    const long long* ei  = (const long long*)d_in[1];  // int64 [2, E]
    const float*     W1  = (const float*)d_in[2];
    const float*     b1  = (const float*)d_in[3];
    const float*     W2  = (const float*)d_in[4];
    const float*     b2  = (const float*)d_in[5];
    float*           out = (float*)d_out;

    const int E = in_sizes[1] / 2;
    const long long* src = ei;
    const long long* dst = ei + E;

    // workspace layout (byte offsets, 256B aligned sections)
    char* ws = (char*)d_ws;
    size_t off = 0;
    float* dis = (float*)(ws + off);  off = align_up(off + sizeof(float) * NNODES, 256);
    __bf16* Wt1 = (__bf16*)(ws + off); off = align_up(off + sizeof(__bf16) * HID * FIN, 256);
    __bf16* Wt2 = (__bf16*)(ws + off); off = align_up(off + sizeof(__bf16) * FOUTP * HID, 256);
    float* xw  = (float*)(ws + off);  off = align_up(off + sizeof(float) * (size_t)NNODES * HID, 256);
    float* agg = (float*)(ws + off);

    // 0) pack weights -> transposed padded bf16 (tiny, once per launch)
    pack_wt_kernel<<<(HID * FIN + 255) / 256, 256, 0, stream>>>(W1, Wt1, FIN, HID, HID);
    pack_wt_kernel<<<(FOUTP * HID + 255) / 256, 256, 0, stream>>>(W2, Wt2, HID, FOUT, FOUTP);

    // 1) degree (with self-loop) -> deg_inv_sqrt
    zero_f32_kernel<<<(NNODES + 255) / 256, 256, 0, stream>>>(dis, NNODES);
    degree_kernel<<<(E + 255) / 256, 256, 0, stream>>>(dst, dis, E);
    rsqrt_kernel<<<(NNODES + 255) / 256, 256, 0, stream>>>(dis, NNODES);

    // 2) xw1 = x @ W1 : [100000,128]@[128,256]  -- WMMA bf16, K fully unrolled
    {
        int tiles = (NNODES / 16) * (HID / 16);          // 6250 * 16
        int blocks = (tiles + 7) / 8;                    // 8 waves/block (256 thr)
        gemm_bf16_wmma_kernel<FIN><<<blocks, 256, 0, stream>>>(x, Wt1, xw, NNODES, HID);
    }

    // 3) agg1 = scatter_add(msg); then h = relu(agg1 + dis^2*xw1 + b1) in place
    {
        long long n = (long long)NNODES * HID;
        zero_f32_kernel<<<(unsigned)((n + 255) / 256), 256, 0, stream>>>(agg, n);
        long long tw = (long long)E * (HID / 4);
        scatter4_kernel<<<(unsigned)((tw + 255) / 256), 256, 0, stream>>>(
            src, dst, dis, xw, HID, agg, HID, HID / 4, E);
        fuse_relu_kernel<<<(unsigned)((n + 255) / 256), 256, 0, stream>>>(
            agg, xw, dis, b1, NNODES, HID);
    }

    // 4) xw2 = h @ W2 : [100000,256]@[256,40->48] -- WMMA bf16, reuse xw buffer
    {
        int tiles = (NNODES / 16) * (FOUTP / 16);        // 6250 * 3
        int blocks = (tiles + 7) / 8;
        gemm_bf16_wmma_kernel<HID><<<blocks, 256, 0, stream>>>(agg, Wt2, xw, NNODES, FOUTP);
    }

    // 5) agg2 = scatter_add (reuse agg buffer with stride 40)
    {
        long long n = (long long)NNODES * FOUT;
        zero_f32_kernel<<<(unsigned)((n + 255) / 256), 256, 0, stream>>>(agg, n);
        long long tw = (long long)E * (FOUT / 4);
        scatter4_kernel<<<(unsigned)((tw + 255) / 256), 256, 0, stream>>>(
            src, dst, dis, xw, FOUTP, agg, FOUT, FOUT / 4, E);
    }

    // 6) out = log_softmax(agg2 + dis^2*xw2 + b2)
    final_kernel<<<(NNODES + 255) / 256, 256, 0, stream>>>(agg, xw, dis, b2, out, NNODES);
}
